// Transformer_12919261626865
// MI455X (gfx1250) — matrix operand order repaired and checked by reading, not compiled, not run on hardware
//
#include <hip/hip_runtime.h>

#define TB 256

typedef __bf16 bf16;
typedef __attribute__((ext_vector_type(16))) __bf16 v16bf;
typedef __attribute__((ext_vector_type(8)))  __bf16 v8bf;
typedef __attribute__((ext_vector_type(8)))  float  v8f;
typedef __attribute__((ext_vector_type(4)))  unsigned int v4u;
typedef __attribute__((ext_vector_type(8)))  int v8i;
typedef __attribute__((ext_vector_type(4)))  int v4i;

// model dims
#define MB  2
#define MT  1024
#define MD  768
#define MH  12
#define MDH 64
#define MF  3072
#define MNB 12
#define MV  50257

// LDS tile row pitch in bf16 elements: 32 data + 8 pad -> 20 dwords stride,
// all 16 fragment rows land in distinct banks for ds_load_b128.
#define LROW 40

enum { GF_BIAS = 1, GF_GELU = 2, GF_RESID = 4, GF_F32OUT = 8, GF_BF16OUT = 16 };

__device__ __forceinline__ bf16 f2bf(float f) {
  union { float f; unsigned u; } c; c.f = f;
  unsigned r = c.u + 0x7FFFu + ((c.u >> 16) & 1u);   // round-to-nearest-even
  unsigned short h = (unsigned short)(r >> 16);
  bf16 b; __builtin_memcpy(&b, &h, 2);
  return b;
}

// Async DMA: 16 bytes global -> LDS, per-lane addresses, tracked by ASYNCcnt.
__device__ __forceinline__ void async_copy_b128(unsigned lds_addr, const void* gptr) {
  asm volatile("global_load_async_to_lds_b128 %0, %1, off"
               :: "v"(lds_addr), "v"(gptr) : "memory");
}
__device__ __forceinline__ void wait_async0() {
  asm volatile("s_wait_asynccnt 0x0" ::: "memory");
}

// Tensor Data Mover: one-instruction 2D tile DMA global -> LDS.
// Loads a (tile_dim1=128 rows) x (tile_dim0=32 bf16) tile with row stride
// `rowStride` (elements).  Rows >= rowsValid read as zero (TDM OOB rule).
// pad_enable inserts 4 dwords of LDS padding after every 16 dwords, producing
// exactly the LROW=40 bank-conflict-free pitch.
__device__ __forceinline__ void tdm_load_2d_tile(
    unsigned ldsAddr, const void* gaddr, unsigned long long rowStride,
    unsigned rowsValid)
{
  unsigned long long ga = (unsigned long long)(uintptr_t)gaddr;
  v4u g0;
  g0[0] = 1u;                                              // count=1 (valid user D#)
  g0[1] = ldsAddr;                                         // lds_addr (bytes)
  g0[2] = (unsigned)ga;                                    // global_addr[31:0]
  g0[3] = (unsigned)((ga >> 32) & 0x01ffffffu)             // global_addr[56:32]
        | 0x80000000u;                                     // type=2 ("image")
  v8i g1;
  g1[0] = (int)((1u << 16)       // data_size = 2 bytes
              | (1u << 20)       // pad_enable
              | (3u << 22)       // pad_interval: 16 dwords
              | (3u << 25));     // pad_amount: 4 dwords
  const unsigned td0 = 32u;      // tensor_dim0 (elements per row from tile start)
  g1[1] = (int)(td0 << 16);                                // tensor_dim0 lo16 @ [31:16]
  g1[2] = (int)((td0 >> 16) | (rowsValid << 16));          // dim0 hi16 | tensor_dim1 lo16
  g1[3] = (int)((rowsValid >> 16) | (32u << 16));          // dim1 hi16 | tile_dim0=32
  g1[4] = (int)128u;                                       // tile_dim1=128, tile_dim2=0
  g1[5] = (int)(unsigned)(rowStride & 0xffffffffu);        // tensor_dim0_stride lo32
  g1[6] = (int)(unsigned)((rowStride >> 32) & 0xffffu);    // dim0_stride hi16
  g1[7] = 0;
  const v4i zero4 = (v4i){0, 0, 0, 0};
  const v8i zero8 = (v8i){0, 0, 0, 0, 0, 0, 0, 0};
  __builtin_amdgcn_tensor_load_to_lds(g0, g1, zero4, zero4, zero8, 0);
}

// ---------------------------------------------------------------------------
// General bf16 WMMA GEMM:  C[m,n] = scale * sum_k A[m,k] * W[n,k]  (+bias,...)
// A: [M,K] row-major (lda), W: [N,K] row-major (ldb)  -> NT layout
// Batched over blockIdx.z with z = z0*zdiv + z1 and per-operand strides.
// 8 waves / block; each wave computes a 32x64 tile (2 M-tiles x 4 N-tiles).
// A tile: per-lane async global->LDS copies (threads 0..127, one 64B row).
// B tile: single TDM tensor_load_to_lds issued by wave 0.
// Both double-buffered; one barrier per K step.
// ---------------------------------------------------------------------------
__global__ __launch_bounds__(TB) void gemm_bf16_wmma(
    const bf16* __restrict__ A, const bf16* __restrict__ Bm,
    const float* __restrict__ bias, const float* resid,
    float* Cf, bf16* Cb,
    int M, int N, int K, int lda, int ldb, int ldc,
    int zdiv,
    long long sA0, long long sA1, long long sB0, long long sB1,
    long long sC0, long long sC1,
    int flags, float scale)
{
  __shared__ bf16 lA[2][128 * LROW];
  __shared__ bf16 lB[2][128 * LROW];

  const int tid  = threadIdx.x;
  const int lane = tid & 31;
  const int wave = tid >> 5;
  const int half = lane >> 4;        // 0: lanes 0-15, 1: lanes 16-31
  const int row  = lane & 15;
  const int wm = wave & 3;           // 4 waves along M
  const int wn = wave >> 2;          // 2 waves along N
  const int mBase = blockIdx.y * 128 + wm * 32;
  const int nBase = blockIdx.x * 128 + wn * 64;

  const int z  = blockIdx.z;
  const int z0 = z / zdiv, z1 = z % zdiv;
  A  += (long long)z0 * sA0 + (long long)z1 * sA1;
  Bm += (long long)z0 * sB0 + (long long)z1 * sB1;
  const long long cOff = (long long)z0 * sC0 + (long long)z1 * sC1;

  // --- A staging: threads 0..127 each own one 64B tile row.
  const bf16* gRowA = A + (long long)(blockIdx.y * 128 + (tid & 127)) * lda;
  unsigned ldsDstA[2];
  ldsDstA[0] = (unsigned)(uintptr_t)&lA[0][(tid & 127) * LROW];
  ldsDstA[1] = (unsigned)(uintptr_t)&lA[1][(tid & 127) * LROW];

  // --- B staging: TDM descriptor basics (wave 0 issues it).
  const int nTile0 = blockIdx.x * 128;
  const unsigned rowsValidB = (unsigned)(N - nTile0);     // rows past this read 0
  const bf16* gTileB = Bm + (long long)nTile0 * ldb;
  unsigned ldsB[2];
  ldsB[0] = (unsigned)(uintptr_t)&lB[0][0];
  ldsB[1] = (unsigned)(uintptr_t)&lB[1][0];

  auto stage = [&](int k0, int bi) {
    if (tid < 128) {
      const bf16* src = gRowA + k0;
#pragma unroll
      for (int c = 0; c < 4; ++c)
        async_copy_b128(ldsDstA[bi] + c * 16, src + c * 8);
    }
    if (wave == 0)
      tdm_load_2d_tile(ldsB[bi], gTileB + k0, (unsigned long long)ldb, rowsValidB);
  };

  v8f acc[2][4];
#pragma unroll
  for (int mt = 0; mt < 2; ++mt)
#pragma unroll
    for (int nt = 0; nt < 4; ++nt)
      acc[mt][nt] = (v8f){0.f, 0.f, 0.f, 0.f, 0.f, 0.f, 0.f, 0.f};

  stage(0, 0);
  int cur = 0;
  for (int k0 = 0; k0 < K; k0 += 32) {
    wait_async0();                                 // this wave's A copies landed
    if (wave == 0) __builtin_amdgcn_s_wait_tensorcnt(0);  // B tile landed
    __syncthreads();                               // publish to all waves
    if (k0 + 32 < K) stage(k0 + 32, cur ^ 1);

    // Fragments from LDS.
    // A 16x32 layout: lanes 0-15 hold K={0..7,16..23}; lanes 16-31 K={8..15,24..31}.
    v16bf aF[2], bF[4];
#pragma unroll
    for (int mt = 0; mt < 2; ++mt) {
      const bf16* p = &lA[cur][(wm * 32 + mt * 16 + row) * LROW + half * 8];
      v8bf lo = *(const v8bf*)(p);
      v8bf hi = *(const v8bf*)(p + 16);
      aF[mt] = __builtin_shufflevector(lo, hi, 0,1,2,3,4,5,6,7,8,9,10,11,12,13,14,15);
    }
    // B: lane = N index; lanes 0-15 hold K={0..7,16..23}; lanes 16-31 K={8..15,24..31}.
#pragma unroll
    for (int nt = 0; nt < 4; ++nt) {
      const bf16* p = &lB[cur][(wn * 64 + nt * 16 + row) * LROW + half * 8];
      v8bf lo = *(const v8bf*)(p);
      v8bf hi = *(const v8bf*)(p + 16);
      bF[nt] = __builtin_shufflevector(lo, hi, 0,1,2,3,4,5,6,7,8,9,10,11,12,13,14,15);
    }
#pragma unroll
    for (int mt = 0; mt < 2; ++mt)
#pragma unroll
      for (int nt = 0; nt < 4; ++nt)
        acc[mt][nt] = __builtin_amdgcn_wmma_f32_16x16x32_bf16(
            false, aF[mt], false, bF[nt], (short)0, acc[mt][nt], false, false);
    cur ^= 1;
  }

  // C/D layout: element i of the v8f sits at (m = i + 8*half, n = lane%16).
#pragma unroll
  for (int mt = 0; mt < 2; ++mt) {
#pragma unroll
    for (int nt = 0; nt < 4; ++nt) {
      int n = nBase + nt * 16 + row;
      if (n >= N) continue;
      float bval = (flags & GF_BIAS) ? bias[n] : 0.f;
#pragma unroll
      for (int i = 0; i < 8; ++i) {
        int m = mBase + mt * 16 + half * 8 + i;
        float v = acc[mt][nt][i] * scale + bval;
        if (flags & GF_GELU) v = 0.5f * v * (1.f + erff(v * 0.70710678118f));
        long long idx = cOff + (long long)m * ldc + n;
        if (flags & GF_RESID) v += resid[idx];
        if (flags & GF_F32OUT) Cf[idx] = v;
        if (flags & GF_BF16OUT) Cb[idx] = f2bf(v);
      }
    }
  }
}

// ---------------------------------------------------------------------------
__global__ __launch_bounds__(TB) void layernorm_kernel(
    float* x, bf16* xb, const float* __restrict__ g, const float* __restrict__ b)
{
  const long long rowi = blockIdx.x;
  float* xr = x + rowi * MD;
  bf16* xbr = xb + rowi * MD;
  __shared__ float red[TB];
  const int tid = threadIdx.x;

  float s = 0.f;
  for (int c = tid; c < MD; c += TB) s += xr[c];
  red[tid] = s; __syncthreads();
  for (int st = TB / 2; st > 0; st >>= 1) { if (tid < st) red[tid] += red[tid + st]; __syncthreads(); }
  float mu = red[0] / MD;
  __syncthreads();

  float vs = 0.f;
  for (int c = tid; c < MD; c += TB) { float d = xr[c] - mu; vs += d * d; }
  red[tid] = vs; __syncthreads();
  for (int st = TB / 2; st > 0; st >>= 1) { if (tid < st) red[tid] += red[tid + st]; __syncthreads(); }
  float rstd = rsqrtf(red[0] / MD + 1e-5f);

  for (int c = tid; c < MD; c += TB) {
    float y = (xr[c] - mu) * rstd * g[c] + b[c];
    xr[c] = y;
    xbr[c] = f2bf(y);
  }
}

__global__ __launch_bounds__(TB) void embed_kernel(
    const int* __restrict__ tokens, const float* __restrict__ WE,
    const float* __restrict__ Wpos, float* __restrict__ x)
{
  const long long rid = blockIdx.x;               // b*T + t
  const int t = (int)(rid % MT);
  const long long tok = tokens[rid];
  for (int c = threadIdx.x; c < MD; c += TB)
    x[rid * MD + c] = WE[tok * MD + c] + Wpos[(long long)t * MD + c];
}

__global__ __launch_bounds__(TB) void f32_to_bf16_kernel(
    const float* __restrict__ in, bf16* __restrict__ out, long long n)
{
  long long i = (long long)blockIdx.x * TB + threadIdx.x;
  const long long stride = (long long)gridDim.x * TB;
  for (; i < n; i += stride) out[i] = f2bf(in[i]);
}

// Vt[b,h,d,t] = Vb[b,t,h*DH+d]   (so PV GEMM is also NT layout)
__global__ __launch_bounds__(TB) void vtranspose_kernel(
    const bf16* __restrict__ Vb, bf16* __restrict__ Vt)
{
  long long i = (long long)blockIdx.x * TB + threadIdx.x;
  const long long total = (long long)MB * MT * MD;
  if (i >= total) return;
  int c = (int)(i % MD);
  long long bt = i / MD;
  int t = (int)(bt % MT);
  long long b = bt / MT;
  int h = c / MDH, d = c % MDH;
  Vt[(((b * MH + h) * (long long)MDH) + d) * MT + t] = Vb[i];
}

// Causal softmax: one block per (b,h,q) row of S (scale already applied).
__global__ __launch_bounds__(TB) void attn_softmax_kernel(
    const float* __restrict__ S, bf16* __restrict__ P)
{
  const long long rid = blockIdx.x;               // over B*H*T
  const int q = (int)(rid % MT);
  const float* s = S + rid * MT;
  bf16* p = P + rid * MT;
  const int len = q + 1;
  __shared__ float red[TB];
  const int tid = threadIdx.x;

  float mx = -3.4e38f;
  for (int c = tid; c < len; c += TB) mx = fmaxf(mx, s[c]);
  red[tid] = mx; __syncthreads();
  for (int st = TB / 2; st > 0; st >>= 1) { if (tid < st) red[tid] = fmaxf(red[tid], red[tid + st]); __syncthreads(); }
  mx = red[0]; __syncthreads();

  float sum = 0.f;
  for (int c = tid; c < len; c += TB) sum += __expf(s[c] - mx);
  red[tid] = sum; __syncthreads();
  for (int st = TB / 2; st > 0; st >>= 1) { if (tid < st) red[tid] += red[tid + st]; __syncthreads(); }
  const float inv = 1.f / red[0];

  for (int c = tid; c < MT; c += TB)
    p[c] = (c < len) ? f2bf(__expf(s[c] - mx) * inv) : f2bf(0.f);
}

// In-place softmax over the vocab dimension of d_out.
__global__ __launch_bounds__(TB) void softmax_rows_kernel(float* X)
{
  const long long rowi = blockIdx.x;
  float* xr = X + rowi * (long long)MV;
  __shared__ float red[TB];
  const int tid = threadIdx.x;

  float mx = -3.4e38f;
  for (int c = tid; c < MV; c += TB) mx = fmaxf(mx, xr[c]);
  red[tid] = mx; __syncthreads();
  for (int st = TB / 2; st > 0; st >>= 1) { if (tid < st) red[tid] = fmaxf(red[tid], red[tid + st]); __syncthreads(); }
  mx = red[0]; __syncthreads();

  float sum = 0.f;
  for (int c = tid; c < MV; c += TB) sum += __expf(xr[c] - mx);
  red[tid] = sum; __syncthreads();
  for (int st = TB / 2; st > 0; st >>= 1) { if (tid < st) red[tid] += red[tid + st]; __syncthreads(); }
  const float inv = 1.f / red[0];

  for (int c = tid; c < MV; c += TB) xr[c] = __expf(xr[c] - mx) * inv;
}

// ---------------------------------------------------------------------------
extern "C" void kernel_launch(void* const* d_in, const int* in_sizes, int n_in,
                              void* d_out, int out_size, void* d_ws, size_t ws_size,
                              hipStream_t stream)
{
  (void)in_sizes; (void)n_in; (void)out_size; (void)ws_size;
  const int*   tokens = (const int*)  d_in[0];
  const float* W_E    = (const float*)d_in[1];
  const float* W_pos  = (const float*)d_in[2];
  const float* WQ     = (const float*)d_in[3];
  const float* WK     = (const float*)d_in[4];
  const float* WV     = (const float*)d_in[5];
  const float* WO     = (const float*)d_in[6];
  const float* ln1_g  = (const float*)d_in[7];
  const float* ln1_b  = (const float*)d_in[8];
  const float* ln2_g  = (const float*)d_in[9];
  const float* ln2_b  = (const float*)d_in[10];
  const float* w1     = (const float*)d_in[11];
  const float* b1     = (const float*)d_in[12];
  const float* w2     = (const float*)d_in[13];
  const float* b2     = (const float*)d_in[14];
  float* out = (float*)d_out;

  char* wsp = (char*)d_ws;
  size_t off = 0;
  auto take = [&](size_t bytes) -> void* {
    void* p = wsp + off;
    off += (bytes + 255) & ~(size_t)255;
    return p;
  };

  const long long DD = (long long)MD * MD, FD = (long long)MF * MD;
  const long long NTOK = (long long)MB * MT;
  bf16* wq_b = (bf16*)take(MNB * DD * 2);
  bf16* wk_b = (bf16*)take(MNB * DD * 2);
  bf16* wv_b = (bf16*)take(MNB * DD * 2);
  bf16* wo_b = (bf16*)take(MNB * DD * 2);
  bf16* w1_b = (bf16*)take(MNB * FD * 2);
  bf16* w2_b = (bf16*)take(MNB * FD * 2);
  bf16* we_b = (bf16*)take((long long)MV * MD * 2);
  float* x   = (float*)take(NTOK * MD * 4);
  bf16* xb   = (bf16*)take(NTOK * MD * 2);
  bf16* qb   = (bf16*)take(NTOK * MD * 2);
  bf16* kb   = (bf16*)take(NTOK * MD * 2);
  bf16* vb   = (bf16*)take(NTOK * MD * 2);
  bf16* vt   = (bf16*)take(NTOK * MD * 2);
  bf16* ob   = (bf16*)take(NTOK * MD * 2);
  bf16* hb   = (bf16*)take(NTOK * MF * 2);
  float* S   = (float*)take((long long)MB * MH * MT * MT * 4);
  bf16* P    = (bf16*)take((long long)MB * MH * MT * MT * 2);

  auto conv = [&](const float* src, bf16* dst, long long n) {
    f32_to_bf16_kernel<<<dim3(2048), dim3(TB), 0, stream>>>(src, dst, n);
  };
  conv(WQ, wq_b, MNB * DD); conv(WK, wk_b, MNB * DD);
  conv(WV, wv_b, MNB * DD); conv(WO, wo_b, MNB * DD);
  conv(w1, w1_b, MNB * FD); conv(w2, w2_b, MNB * FD);
  conv(W_E, we_b, (long long)MV * MD);

  embed_kernel<<<dim3((unsigned)NTOK), dim3(TB), 0, stream>>>(tokens, W_E, W_pos, x);

  auto gemm = [&](const bf16* A, const bf16* Bmat, const float* bias, const float* resid,
                  float* Cf, bf16* Cb, int M, int N, int K, int lda, int ldb, int ldc,
                  int Z, int zdiv,
                  long long sA0, long long sA1, long long sB0, long long sB1,
                  long long sC0, long long sC1, int flags, float scale) {
    dim3 g((N + 127) / 128, (M + 127) / 128, Z);
    gemm_bf16_wmma<<<g, dim3(TB), 0, stream>>>(A, Bmat, bias, resid, Cf, Cb,
                                               M, N, K, lda, ldb, ldc, zdiv,
                                               sA0, sA1, sB0, sB1, sC0, sC1, flags, scale);
  };

  const int Mtok = (int)NTOK;
  const long long vtr_blocks = (NTOK * MD + TB - 1) / TB;

  for (int i = 0; i < MNB; ++i) {
    layernorm_kernel<<<dim3(Mtok), dim3(TB), 0, stream>>>(
        x, xb, ln1_g + (long long)i * MD, ln1_b + (long long)i * MD);

    gemm(xb, wq_b + i * DD, nullptr, nullptr, nullptr, qb, Mtok, MD, MD, MD, MD, MD,
         1, 1, 0, 0, 0, 0, 0, 0, GF_BF16OUT, 1.f);
    gemm(xb, wk_b + i * DD, nullptr, nullptr, nullptr, kb, Mtok, MD, MD, MD, MD, MD,
         1, 1, 0, 0, 0, 0, 0, 0, GF_BF16OUT, 1.f);
    gemm(xb, wv_b + i * DD, nullptr, nullptr, nullptr, vb, Mtok, MD, MD, MD, MD, MD,
         1, 1, 0, 0, 0, 0, 0, 0, GF_BF16OUT, 1.f);

    vtranspose_kernel<<<dim3((unsigned)vtr_blocks), dim3(TB), 0, stream>>>(vb, vt);

    // scores[b,h,q,k] = (1/8) * Q . K   (batched over z = b*H + h)
    gemm(qb, kb, nullptr, nullptr, S, nullptr, MT, MT, MDH, MD, MD, MT,
         MB * MH, MH, (long long)MT * MD, MDH, (long long)MT * MD, MDH,
         (long long)MH * MT * MT, (long long)MT * MT, GF_F32OUT, 0.125f);

    attn_softmax_kernel<<<dim3(MB * MH * MT), dim3(TB), 0, stream>>>(S, P);

    // O[b,q,h*DH+d] = P @ V
    gemm(P, vt, nullptr, nullptr, nullptr, ob, MT, MDH, MT, MT, MT, MD,
         MB * MH, MH, (long long)MH * MT * MT, (long long)MT * MT,
         (long long)MH * MDH * MT, (long long)MDH * MT,
         (long long)MT * MD, MDH, GF_BF16OUT, 1.f);

    // x += O @ WO^T
    gemm(ob, wo_b + i * DD, nullptr, x, x, nullptr, Mtok, MD, MD, MD, MD, MD,
         1, 1, 0, 0, 0, 0, 0, 0, GF_RESID | GF_F32OUT, 1.f);

    layernorm_kernel<<<dim3(Mtok), dim3(TB), 0, stream>>>(
        x, xb, ln2_g + (long long)i * MD, ln2_b + (long long)i * MD);

    // h = gelu(x @ w1^T + b1)
    gemm(xb, w1_b + i * FD, b1 + (long long)i * MF, nullptr, nullptr, hb,
         Mtok, MF, MD, MD, MD, MF,
         1, 1, 0, 0, 0, 0, 0, 0, GF_BIAS | GF_GELU | GF_BF16OUT, 1.f);

    // x += h @ w2^T + b2
    gemm(hb, w2_b + i * FD, b2 + (long long)i * MD, x, x, nullptr,
         Mtok, MD, MF, MF, MF, MD,
         1, 1, 0, 0, 0, 0, 0, 0, GF_BIAS | GF_RESID | GF_F32OUT, 1.f);
  }

  // logits = x @ W_E^T  -> d_out, then in-place vocab softmax
  conv(x, xb, NTOK * MD);
  gemm(xb, we_b, nullptr, nullptr, out, nullptr, Mtok, MV, MD, MD, MD, MV,
       1, 1, 0, 0, 0, 0, 0, 0, GF_F32OUT, 1.f);
  softmax_rows_kernel<<<dim3(Mtok), dim3(TB), 0, stream>>>(out);
}
